// DecoderV1_75694503624763
// MI455X (gfx1250) — matrix operand order repaired and checked by reading, not compiled, not run on hardware
//
#include <hip/hip_runtime.h>
#include <hip/hip_bf16.h>
#include <math.h>

// ---------------------------------------------------------------------------
// WaveNet-style decoder for MI455X (gfx1250), wave32 + WMMA bf16.
// Grid: 32 blocks x 128 threads (4 waves). Each block owns 64 batch rows,
// runs all 24 autoregressive steps. Dilated state history round-trips through
// d_ws as bf16; weights are pre-transposed (N-major) bf16 in d_ws and
// streamed as WMMA B-fragments (compiler register-caches them across steps).
// Next-level state is prefetched (global_prefetch_b8) to hide the only
// global latency on the sequential dependence chain.
// ---------------------------------------------------------------------------

typedef __attribute__((ext_vector_type(16))) __bf16 v16bf;
typedef __attribute__((ext_vector_type(8)))  float  v8f;

#define NB    2048
#define TPRED 24
#define TENC  168
#define NF    64
#define NFEAT 15
#define BM    64     // batch rows per block
#define NTHR  128    // 4 waves

// packed transposed-weight layout inside d_ws (element offsets, bf16)
#define W1T_OFF   0                 // [64 n][32 k]   (k>=16 zero padded)
#define WCAT_OFF  2048              // [128 n][128 k] (k<64 -> W2, k>=64 -> W3)
#define W4T_OFF   (2048 + 16384)    // [128 n][64 k]
#define W5T_OFF   (W4T_OFF + 8192)  // [128 n][384 k]
#define WT_TOTAL  (W5T_OFF + 49152) // 75776 bf16
#define HIST_ELEMS (5 * TPRED * NB * NF)   // levels 0..4 only (d=32 never read)

union FragU { v16bf v; uint4 q[2]; };

// A-matrix 16x32 bf16 fragment (ISA 7.12.2): lane holds row (lane&15);
// lanes 0-15: K 0..7 & 16..23, lanes 16-31: K 8..15 & 24..31.
__device__ __forceinline__ v16bf fragA(const __bf16* base, int stride, int lane, int koff) {
  const __bf16* p = base + (lane & 15) * stride + koff + ((lane >> 4) & 1) * 8;
  FragU u;
  u.q[0] = *reinterpret_cast<const uint4*>(p);
  u.q[1] = *reinterpret_cast<const uint4*>(p + 16);
  return u.v;
}

// B-matrix 32x16 bf16 fragment from N-major (transposed) storage:
// lane holds col (lane&15); lanes 0-15: K 0..7 & 16..23, lanes 16-31: K 8..15 & 24..31.
__device__ __forceinline__ v16bf fragB(const __bf16* baseT, int stride, int lane, int koff) {
  const __bf16* p = baseT + (lane & 15) * stride + koff + ((lane >> 4) & 1) * 8;
  FragU u;
  u.q[0] = *reinterpret_cast<const uint4*>(p);
  u.q[1] = *reinterpret_cast<const uint4*>(p + 16);
  return u.v;
}

__device__ __forceinline__ v8f wmma_bf16(v16bf a, v16bf b, v8f c) {
  return __builtin_amdgcn_wmma_f32_16x16x32_bf16(false, a, false, b, (short)0, c, false, false);
}

// --- branchless transcendentals using gfx1250 HW TRANS ops -----------------
__device__ __forceinline__ float hw_exp2(float x) {
#if defined(__AMDGCN__) && __has_builtin(__builtin_amdgcn_exp2f)
  return __builtin_amdgcn_exp2f(x);
#else
  return exp2f(x);
#endif
}
__device__ __forceinline__ float hw_rcp(float x) {
#if defined(__AMDGCN__) && __has_builtin(__builtin_amdgcn_rcpf)
  return __builtin_amdgcn_rcpf(x);
#else
  return 1.0f / x;
#endif
}
__device__ __forceinline__ float fast_tanh(float x) {
#if defined(__AMDGCN__) && __has_builtin(__builtin_amdgcn_tanhf)
  return __builtin_amdgcn_tanhf(x);           // v_tanh_f32 (gfx1250 TRANS op)
#else
  float e = hw_exp2(x * 2.8853900817779268f);
  return 1.0f - 2.0f * hw_rcp(e + 1.0f);
#endif
}
__device__ __forceinline__ float fast_sigmoid(float g) {
  return hw_rcp(1.0f + hw_exp2(g * -1.4426950408889634f));
}

// ---------------------------------------------------------------------------
// Prep kernel: f32 weights -> bf16, transposed to N-major for B-fragments.
// ---------------------------------------------------------------------------
__global__ void decoder_prep_weights(const float* __restrict__ W1,
                                     const float* __restrict__ W2,
                                     const float* __restrict__ W3,
                                     const float* __restrict__ W4,
                                     const float* __restrict__ W5,
                                     __bf16* __restrict__ wT) {
  int i = blockIdx.x * blockDim.x + threadIdx.x;
  if (i >= WT_TOTAL) return;
  float v;
  if (i < WCAT_OFF) {                       // W1T [64][32], K padded to 32
    int n = i >> 5, k = i & 31;
    v = (k < 16) ? W1[k * 64 + n] : 0.0f;
  } else if (i < W4T_OFF) {                 // WcatT [128][128] = [W2;W3]^T
    int j = i - WCAT_OFF, n = j >> 7, k = j & 127;
    v = (k < 64) ? W2[k * 128 + n] : W3[(k - 64) * 128 + n];
  } else if (i < W5T_OFF) {                 // W4T [128][64]
    int j = i - W4T_OFF, n = j >> 6, k = j & 63;
    v = W4[k * 128 + n];
  } else {                                  // W5T [128][384]
    int j = i - W5T_OFF, n = j / 384, k = j % 384;
    v = W5[k * 128 + n];
  }
  wT[i] = (__bf16)v;
}

// ---------------------------------------------------------------------------
// Main persistent decoder kernel.
// ---------------------------------------------------------------------------
struct __align__(16) SMemT {
  __bf16 in16[BM * 32];    // [init | feat | 0-pad] per step, K padded to 32
  __bf16 Acat[BM * 128];   // [state(64) | x(64)] bf16
  __bf16 gbuf[BM * 64];    // gated / relu(skips) staging
  float  sb1[64];
  float  sb2[128];
  float  sb4[128];
  float  sb5[128];
  float  sw6[128];
  float  sb6;
};

__global__ __launch_bounds__(NTHR, 1)
void decoder_main(const float* __restrict__ feat,   // [B,24,15]
                  const float* __restrict__ init,   // [B,1]
                  const float* __restrict__ enc,    // [6,B,168,64]
                  const float* __restrict__ b1, const float* __restrict__ b2,
                  const float* __restrict__ b4, const float* __restrict__ b5,
                  const float* __restrict__ W6, const float* __restrict__ b6,
                  const __bf16* __restrict__ wT,
                  __bf16* __restrict__ hist,        // [5,24,B,64] bf16
                  float* __restrict__ out)          // [B,24]
{
  __shared__ SMemT sm;

  const int tid  = threadIdx.x;
  const int lane = tid & 31;
  const int wv   = tid >> 5;        // wave 0..3
  const int l15  = lane & 15;
  const int half = (lane >> 4) & 1;
  const int wrow = wv * 16;         // wave's row base within the 64-row tile
  const int gb   = blockIdx.x * BM; // global batch base

  const __bf16* W1T   = wT + W1T_OFF;
  const __bf16* WcatT = wT + WCAT_OFF;
  const __bf16* W4T   = wT + W4T_OFF;
  const __bf16* W5T   = wT + W5T_OFF;

  for (int i = tid; i < 64;  i += NTHR) sm.sb1[i] = b1[i];
  for (int i = tid; i < 128; i += NTHR) {
    sm.sb2[i] = b2[i]; sm.sb4[i] = b4[i]; sm.sb5[i] = b5[i]; sm.sw6[i] = W6[i];
  }
  if (tid == 0) sm.sb6 = b6[0];

  float xacc[4][8];   // x in WMMA C layout: tile j covers cols 16j..16j+15
  v8f   hacc[8];      // h accumulator across levels, C layout

  for (int t = 0; t < TPRED; ++t) {
    // ---- phase 1: build this_input = [init | feat[:,t]] as bf16, K-pad to 32
    for (int i = tid; i < BM * 32; i += NTHR) {
      int r = i >> 5, c = i & 31;
      float v = 0.0f;
      if (c == 0)            v = init[gb + r];
      else if (c <= NFEAT)   v = feat[((gb + r) * TPRED + t) * NFEAT + (c - 1)];
      sm.in16[i] = (__bf16)v;
    }
    // prefetch next step's feature rows into WGP$ while this step computes
    if (t + 1 < TPRED && tid < BM)
      __builtin_prefetch(&feat[((gb + tid) * TPRED + (t + 1)) * NFEAT], 0, 3);
    __syncthreads();

    // ---- phase 2: x = tanh(in16 @ W1 + b1)   (1 K-step, 4 N-tiles)
    {
      v16bf a = fragA(sm.in16 + wrow * 32, 32, lane, 0);
#pragma unroll
      for (int j = 0; j < 4; ++j) {
        v8f c;
#pragma unroll
        for (int r = 0; r < 8; ++r) c[r] = 0.0f;
        c = wmma_bf16(a, fragB(W1T + (j * 16) * 32, 32, lane, 0), c);
        float bb = sm.sb1[j * 16 + l15];
#pragma unroll
        for (int r = 0; r < 8; ++r) xacc[j][r] = fast_tanh(c[r] + bb);
      }
#pragma unroll
      for (int j = 0; j < 8; ++j)
#pragma unroll
        for (int r = 0; r < 8; ++r) hacc[j][r] = 0.0f;
    }

    // ---- dilation levels (strictly sequential: x carries residuals)
    for (int l = 0; l < 6; ++l) {
      const int d = 1 << l;

      // prefetch NEXT level's dilated state (independent of this level's math)
      {
        const int ln = l + 1, dn = d << 1;
        if (ln < 6) {
          if (t >= dn) {
            const __bf16* p = hist + ((ln * TPRED + (t - dn)) * NB + gb) * NF;
            __builtin_prefetch(p + tid * 32, 0, 3);             // 128 x 64B = 8KB
          } else {
            const float* p = enc + (((size_t)ln * NB + gb) * TENC + (TENC + t - dn)) * NF;
            __builtin_prefetch(p + (size_t)(tid >> 1) * (TENC * NF) + (tid & 1) * 32, 0, 3);
          }
        }
      }

      // 3a: stage current x (pre-residual for this level) into Acat cols 64..127
#pragma unroll
      for (int j = 0; j < 4; ++j)
#pragma unroll
        for (int r = 0; r < 8; ++r) {
          int m = wrow + r + 8 * half;
          sm.Acat[m * 128 + 64 + j * 16 + l15] = (__bf16)xacc[j][r];
        }

      // 3b: load dilated state into Acat cols 0..63 (uniform branch hoisted)
      if (t >= d) {  // implies l<=4 since d=32 > 23
        const __bf16* hsrc = hist + ((l * TPRED + (t - d)) * NB + gb) * NF;
        for (int i = tid; i < BM * NF; i += NTHR) {
          int r = i >> 6, c = i & 63;
          sm.Acat[r * 128 + c] = hsrc[i];
        }
      } else {
        const float* esrc = enc + (((size_t)l * NB + gb) * TENC + (TENC + t - d)) * NF;
        for (int i = tid; i < BM * NF; i += NTHR) {
          int r = i >> 6, c = i & 63;
          sm.Acat[r * 128 + c] = (__bf16)esrc[(size_t)r * (TENC * NF) + c];
        }
      }
      __syncthreads();

      // 3c+3d: dc = [state|x] @ [W2;W3]; gate pairs (j, j+4) to cap live accs
      v16bf af[4];
#pragma unroll
      for (int k = 0; k < 4; ++k) af[k] = fragA(sm.Acat + wrow * 128, 128, lane, k * 32);
#pragma unroll
      for (int j = 0; j < 4; ++j) {
        v8f accF, accG;
#pragma unroll
        for (int r = 0; r < 8; ++r) { accF[r] = 0.0f; accG[r] = 0.0f; }
#pragma unroll
        for (int k = 0; k < 4; ++k)
          accF = wmma_bf16(af[k], fragB(WcatT + (j * 16) * 128, 128, lane, k * 32), accF);
#pragma unroll
        for (int k = 0; k < 4; ++k)
          accG = wmma_bf16(af[k], fragB(WcatT + ((j + 4) * 16) * 128, 128, lane, k * 32), accG);
        float bf_ = sm.sb2[j * 16 + l15];
        float bg_ = sm.sb2[64 + j * 16 + l15];
#pragma unroll
        for (int r = 0; r < 8; ++r) {
          float gt = fast_tanh(accF[r] + bf_) * fast_sigmoid(accG[r] + bg_);
          sm.gbuf[(wrow + r + 8 * half) * 64 + j * 16 + l15] = (__bf16)gt;
        }
      }
      __syncthreads();

      // 3e+3f: out = gated @ W4; pairs (skip j, res j+4); relu/res/x update
      v16bf ag0 = fragA(sm.gbuf + wrow * 64, 64, lane, 0);
      v16bf ag1 = fragA(sm.gbuf + wrow * 64, 64, lane, 32);
      float skipv[4][8];
#pragma unroll
      for (int j = 0; j < 4; ++j) {
        v8f ocS, ocR;
#pragma unroll
        for (int r = 0; r < 8; ++r) { ocS[r] = 0.0f; ocR[r] = 0.0f; }
        ocS = wmma_bf16(ag0, fragB(W4T + (j * 16) * 64, 64, lane, 0),  ocS);
        ocS = wmma_bf16(ag1, fragB(W4T + (j * 16) * 64, 64, lane, 32), ocS);
        ocR = wmma_bf16(ag0, fragB(W4T + ((j + 4) * 16) * 64, 64, lane, 0),  ocR);
        ocR = wmma_bf16(ag1, fragB(W4T + ((j + 4) * 16) * 64, 64, lane, 32), ocR);
        float bs = sm.sb4[j * 16 + l15];
        float br = sm.sb4[64 + j * 16 + l15];
#pragma unroll
        for (int r = 0; r < 8; ++r) {
          skipv[j][r] = fmaxf(0.0f, ocS[r] + bs);
          xacc[j][r] += ocR[r] + br;
        }
      }
      __syncthreads();   // gbuf fragA reads (all waves) done before overwrite
#pragma unroll
      for (int j = 0; j < 4; ++j)
#pragma unroll
        for (int r = 0; r < 8; ++r)
          sm.gbuf[(wrow + r + 8 * half) * 64 + j * 16 + l15] = (__bf16)skipv[j][r];
      if (l < 5 && (t + d) < TPRED) {   // only store history that will be read
#pragma unroll
        for (int j = 0; j < 4; ++j)
#pragma unroll
          for (int r = 0; r < 8; ++r) {
            int b = gb + wrow + r + 8 * half;
            hist[((l * TPRED + t) * NB + b) * NF + j * 16 + l15] = (__bf16)xacc[j][r];
          }
        __threadfence_block();
      }
      __syncthreads();

      // 3g: hacc += relu(skips_l) @ W5[64l:64l+64, :]   (2 k-steps, 8 N-tiles)
      v16bf as0 = fragA(sm.gbuf + wrow * 64, 64, lane, 0);
      v16bf as1 = fragA(sm.gbuf + wrow * 64, 64, lane, 32);
#pragma unroll
      for (int j = 0; j < 8; ++j) {
        hacc[j] = wmma_bf16(as0, fragB(W5T + (j * 16) * 384, 384, lane, l * 64),      hacc[j]);
        hacc[j] = wmma_bf16(as1, fragB(W5T + (j * 16) * 384, 384, lane, l * 64 + 32), hacc[j]);
      }
    } // levels

    // ---- phase 4: h = relu(hacc + b5); y = h @ W6 + b6  (16-lane reduction)
    {
      float part[8];
#pragma unroll
      for (int r = 0; r < 8; ++r) part[r] = 0.0f;
#pragma unroll
      for (int j = 0; j < 8; ++j) {
        float b5v = sm.sb5[j * 16 + l15];
        float w6v = sm.sw6[j * 16 + l15];
#pragma unroll
        for (int r = 0; r < 8; ++r) {
          float hv = fmaxf(0.0f, hacc[j][r] + b5v);
          part[r] += hv * w6v;
        }
      }
#pragma unroll
      for (int m = 1; m < 16; m <<= 1)
#pragma unroll
        for (int r = 0; r < 8; ++r) part[r] += __shfl_xor(part[r], m, 32);
      if (l15 == 0) {
#pragma unroll
        for (int r = 0; r < 8; ++r) {
          int b = gb + wrow + r + 8 * half;
          out[b * TPRED + t] = part[r] + sm.sb6;
        }
      }
    }
    __syncthreads();
  } // t
}

// ---------------------------------------------------------------------------
extern "C" void kernel_launch(void* const* d_in, const int* in_sizes, int n_in,
                              void* d_out, int out_size, void* d_ws, size_t ws_size,
                              hipStream_t stream) {
  (void)in_sizes; (void)n_in; (void)out_size; (void)ws_size;
  const float* feat = (const float*)d_in[0];   // decoder_features [B,24,15]
  const float* init = (const float*)d_in[1];   // decoder_init_input [B,1]
  const float* enc  = (const float*)d_in[2];   // encoder_outputs [6,B,168,64]
  const float* W1   = (const float*)d_in[3];
  const float* b1   = (const float*)d_in[4];
  const float* W2   = (const float*)d_in[5];
  const float* b2   = (const float*)d_in[6];
  const float* W3   = (const float*)d_in[7];
  const float* W4   = (const float*)d_in[8];
  const float* b4   = (const float*)d_in[9];
  const float* W5   = (const float*)d_in[10];
  const float* b5   = (const float*)d_in[11];
  const float* W6   = (const float*)d_in[12];
  const float* b6   = (const float*)d_in[13];

  __bf16* wsb  = reinterpret_cast<__bf16*>(d_ws);
  __bf16* hist = wsb;                 // 5*24*2048*64 bf16  (~31.5 MB)
  __bf16* wT   = wsb + HIST_ELEMS;    // 75776 bf16 transposed weights

  decoder_prep_weights<<<(WT_TOTAL + 255) / 256, 256, 0, stream>>>(W1, W2, W3, W4, W5, wT);
  decoder_main<<<NB / BM, NTHR, 0, stream>>>(feat, init, enc, b1, b2, b4, b5, W6, b6,
                                             wT, hist, (float*)d_out);
}
